// HeteroGATv2_24696061952070
// MI455X (gfx1250) — compile-verified, matrix operand order checked by reading, not run on hardware
//
#include <hip/hip_runtime.h>

// ---------------------------------------------------------------------------
// HeteroGATv2 forward for MI455X (gfx1250, wave32).
//  - Node GEMMs (x@Wl+bl, x@Wr+br) via v_wmma_f32_16x16x32_bf16 with
//    split-bf16 (hi/lo) 3-product emulation for near-fp32 accuracy.
//  - Edge phase: wave-per-edge coalesced gathers, segment-max via ordered-uint
//    atomicMax, exp+weighted scatter via float atomics, normalization folded
//    into a node pass (sum(alpha*xl) == sum(ex*xl)/sum(ex)).
// ---------------------------------------------------------------------------

typedef __attribute__((ext_vector_type(16))) __bf16 v16bf;
typedef __attribute__((ext_vector_type(8)))  float  v8f;

#define NNODES 50000
#define EDGES  800000
#define HID    64
#define DIN    32
#define NELEM  (NNODES * HID)

// ---- ordered-uint encoding for atomic float max --------------------------
__device__ __forceinline__ unsigned f2ord(float f) {
  unsigned u = __float_as_uint(f);
  return (u & 0x80000000u) ? ~u : (u | 0x80000000u);
}
__device__ __forceinline__ float ord2f(unsigned k) {
  unsigned u = (k & 0x80000000u) ? (k & 0x7fffffffu) : ~k;
  return __uint_as_float(u);
}

// ---------------------------------------------------------------------------
// Y[N,64] = X[N,K] @ W[K,64] + bias, K in {32,64}, N = 50000 (multiple of 16).
// Block = 128 threads = 4 waves; wave w owns the 16-col tile w of a 16-row
// tile. Split-bf16: X = Xh + Xl, W = Wh + Wl;  Y ~= Xh*Wh + Xh*Wl + Xl*Wh.
// A frag layout (ISA 7.12.2, 16-bit A 16x32): lane%16 = M; slot v, pair p:
//   K = (v<4 ? 2v : 2v+8) + (lane>=16 ? 8 : 0) + p.  B mirrored (lane%16 = N).
// C/D: vgpr r, lane l -> M = r + 8*(l>=16), N = l%16.
// ---------------------------------------------------------------------------
__global__ __launch_bounds__(128)
void gemm_node_kernel(const float* __restrict__ X, const float* __restrict__ W,
                      const float* __restrict__ bias, float* __restrict__ Y,
                      int K) {
  const int lane   = threadIdx.x & 31;
  const int wave   = threadIdx.x >> 5;             // 0..3 : 16-col tile
  const int rt     = blockIdx.x;                   // 16-row tile (3125)
  const int m16    = lane & 15;
  const int row    = rt * 16 + m16;
  const int ncol   = wave * 16 + m16;
  const int hiHalf = (lane >= 16) ? 8 : 0;

  v8f c;
  const float bv = bias[ncol];
#pragma unroll
  for (int r = 0; r < 8; ++r) c[r] = bv;

  for (int kk = 0; kk < K; kk += 32) {
    v16bf ahi, alo, bhi, blo;
#pragma unroll
    for (int v = 0; v < 8; ++v) {
      const int kb = kk + ((v < 4) ? 2 * v : 2 * v + 8) + hiHalf;
      const float a0 = X[row * K + kb];
      const float a1 = X[row * K + kb + 1];
      const float b0 = W[kb * HID + ncol];
      const float b1 = W[(kb + 1) * HID + ncol];
      const __bf16 a0h = (__bf16)a0, a1h = (__bf16)a1;
      const __bf16 b0h = (__bf16)b0, b1h = (__bf16)b1;
      ahi[2 * v] = a0h;                       ahi[2 * v + 1] = a1h;
      alo[2 * v] = (__bf16)(a0 - (float)a0h); alo[2 * v + 1] = (__bf16)(a1 - (float)a1h);
      bhi[2 * v] = b0h;                       bhi[2 * v + 1] = b1h;
      blo[2 * v] = (__bf16)(b0 - (float)b0h); blo[2 * v + 1] = (__bf16)(b1 - (float)b1h);
    }
    c = __builtin_amdgcn_wmma_f32_16x16x32_bf16(false, ahi, false, bhi, (short)0, c, false, false);
    c = __builtin_amdgcn_wmma_f32_16x16x32_bf16(false, alo, false, bhi, (short)0, c, false, false);
    c = __builtin_amdgcn_wmma_f32_16x16x32_bf16(false, ahi, false, blo, (short)0, c, false, false);
  }

#pragma unroll
  for (int r = 0; r < 8; ++r)
    Y[(rt * 16 + r + hiHalf) * HID + ncol] = c[r];
}

// ---- per-edge-type segment state init -------------------------------------
__global__ __launch_bounds__(256)
void seg_init_kernel(unsigned* __restrict__ menc, float* __restrict__ s,
                     float* __restrict__ acc) {
  const int idx = blockIdx.x * blockDim.x + threadIdx.x;
  if (idx < NELEM) acc[idx] = 0.f;
  if (idx < NNODES) { menc[idx] = f2ord(-__builtin_inff()); s[idx] = 0.f; }
}

// ---- edge pass 1: e = att . leakyrelu(xl[src] + xr[dst]); segment max -----
__global__ __launch_bounds__(256)
void edge_attn_kernel(const int* __restrict__ esrc, const int* __restrict__ edst,
                      const float* __restrict__ xl, const float* __restrict__ xr,
                      const float* __restrict__ att, float* __restrict__ e,
                      unsigned* __restrict__ menc) {
  const int gid  = blockIdx.x * blockDim.x + threadIdx.x;
  const int eidx = gid >> 5;
  const int lane = gid & 31;
  if (eidx >= EDGES) return;
  const int si = esrc[eidx];
  const int di = edst[eidx];
  const float2 a = *(const float2*)(xl + si * HID + 2 * lane);
  const float2 b = *(const float2*)(xr + di * HID + 2 * lane);
  float v0 = a.x + b.x; v0 = (v0 > 0.f) ? v0 : 0.2f * v0;
  float v1 = a.y + b.y; v1 = (v1 > 0.f) ? v1 : 0.2f * v1;
  float p = v0 * att[2 * lane] + v1 * att[2 * lane + 1];
#pragma unroll
  for (int off = 16; off > 0; off >>= 1) p += __shfl_xor(p, off, 32);
  if (lane == 0) {
    e[eidx] = p;
    atomicMax(&menc[di], f2ord(p));
  }
}

// ---- edge pass 2: ex = exp(e - m[dst]); s[dst]+=ex; acc[dst]+=ex*xl[src] --
__global__ __launch_bounds__(256)
void edge_scatter_kernel(const int* __restrict__ esrc, const int* __restrict__ edst,
                         const float* __restrict__ xl, const float* __restrict__ e,
                         const unsigned* __restrict__ menc,
                         float* __restrict__ s, float* __restrict__ acc) {
  const int gid  = blockIdx.x * blockDim.x + threadIdx.x;
  const int eidx = gid >> 5;
  const int lane = gid & 31;
  if (eidx >= EDGES) return;
  const int si = esrc[eidx];
  const int di = edst[eidx];
  const float ex = __expf(e[eidx] - ord2f(menc[di]));
  const float2 a = *(const float2*)(xl + si * HID + 2 * lane);
  atomicAdd(acc + di * HID + 2 * lane,     ex * a.x);
  atomicAdd(acc + di * HID + 2 * lane + 1, ex * a.y);
  if (lane == 0) atomicAdd(s + di, ex);
}

// ---- node pass: out += acc/(s+eps) + bias ---------------------------------
__global__ __launch_bounds__(256)
void node_accum_kernel(const float* __restrict__ acc, const float* __restrict__ s,
                       const float* __restrict__ bias, float* __restrict__ out) {
  const int idx = blockIdx.x * blockDim.x + threadIdx.x;
  if (idx >= NELEM) return;
  const int i = idx >> 6;
  const int k = idx & 63;
  out[idx] += acc[idx] / (s[i] + 1e-16f) + bias[k];
}

__global__ __launch_bounds__(256)
void zero_kernel(float* __restrict__ p, int n) {
  const int i = blockIdx.x * blockDim.x + threadIdx.x;
  if (i < n) p[i] = 0.f;
}

__global__ __launch_bounds__(256)
void relu_kernel(float* __restrict__ p, int n) {
  const int i = blockIdx.x * blockDim.x + threadIdx.x;
  if (i < n) p[i] = fmaxf(p[i], 0.f);
}

// ---- output head: softmax(h0 @ Wout + bout) -------------------------------
__global__ __launch_bounds__(256)
void head_kernel(const float* __restrict__ h, const float* __restrict__ Wout,
                 const float* __restrict__ bout, float* __restrict__ out) {
  const int i = blockIdx.x * blockDim.x + threadIdx.x;
  if (i >= NNODES) return;
  float l[4] = {bout[0], bout[1], bout[2], bout[3]};
  for (int k = 0; k < HID; ++k) {
    const float hv = h[i * HID + k];
#pragma unroll
    for (int c = 0; c < 4; ++c) l[c] += hv * Wout[k * 4 + c];
  }
  const float m = fmaxf(fmaxf(l[0], l[1]), fmaxf(l[2], l[3]));
  float ex[4], ssum = 0.f;
#pragma unroll
  for (int c = 0; c < 4; ++c) { ex[c] = __expf(l[c] - m); ssum += ex[c]; }
  const float inv = 1.f / ssum;
#pragma unroll
  for (int c = 0; c < 4; ++c) out[i * 4 + c] = ex[c] * inv;
}

// ---------------------------------------------------------------------------
extern "C" void kernel_launch(void* const* d_in, const int* in_sizes, int n_in,
                              void* d_out, int out_size, void* d_ws, size_t ws_size,
                              hipStream_t stream) {
  const float* x     = (const float*)d_in[0];
  const int*   edges = (const int*)d_in[1];
  const float* Wl0   = (const float*)d_in[2];
  const float* Wr0   = (const float*)d_in[3];
  const float* bl0   = (const float*)d_in[4];
  const float* br0   = (const float*)d_in[5];
  const float* att0  = (const float*)d_in[6];
  const float* bias0 = (const float*)d_in[7];
  const float* Wl1   = (const float*)d_in[8];
  const float* Wr1   = (const float*)d_in[9];
  const float* bl1   = (const float*)d_in[10];
  const float* br1   = (const float*)d_in[11];
  const float* att1  = (const float*)d_in[12];
  const float* bias1 = (const float*)d_in[13];
  const float* Wout  = (const float*)d_in[14];
  const float* bout  = (const float*)d_in[15];
  float* out = (float*)d_out;
  char*  ws  = (char*)d_ws;

  // workspace layout (bytes)
  float*    hA   = (float*)(ws);                    // 3*N*H = 38.4 MB
  float*    hB   = (float*)(ws + 38400000);         // 3*N*H
  float*    xl   = (float*)(ws + 76800000);         // N*H
  float*    xr   = (float*)(ws + 89600000);         // N*H
  float*    ebuf = (float*)(ws + 102400000);        // E
  unsigned* menc = (unsigned*)(ws + 105600000);     // N
  float*    sbuf = (float*)(ws + 105800000);        // N
  float*    acc  = (float*)(ws + 106000000);        // N*H

  static const int SRCt[7] = {0, 1, 1, 1, 2, 2, 2};
  static const int DSTt[7] = {0, 0, 1, 2, 0, 1, 2};

  const dim3 b256(256);
  const int gNode = (NELEM + 255) / 256;        // 12500
  const int gAll3 = (3 * NELEM + 255) / 256;    // 37500
  const int gEdge = (EDGES * 32) / 256;         // 100000
  const int gGemm = NNODES / 16;                // 3125

  // ---------------- layer 0 (D_IN=32 -> H=64) ----------------
  zero_kernel<<<gAll3, b256, 0, stream>>>(hA, 3 * NELEM);
  for (int t = 0; t < 7; ++t) {
    const int* esrc = edges + t * 2 * EDGES;
    const int* edst = esrc + EDGES;
    gemm_node_kernel<<<gGemm, 128, 0, stream>>>(
        x + SRCt[t] * NNODES * DIN, Wl0 + t * DIN * HID, bl0 + t * HID, xl, DIN);
    gemm_node_kernel<<<gGemm, 128, 0, stream>>>(
        x + DSTt[t] * NNODES * DIN, Wr0 + t * DIN * HID, br0 + t * HID, xr, DIN);
    seg_init_kernel<<<gNode, b256, 0, stream>>>(menc, sbuf, acc);
    edge_attn_kernel<<<gEdge, b256, 0, stream>>>(esrc, edst, xl, xr,
                                                 att0 + t * HID, ebuf, menc);
    edge_scatter_kernel<<<gEdge, b256, 0, stream>>>(esrc, edst, xl, ebuf, menc,
                                                    sbuf, acc);
    node_accum_kernel<<<gNode, b256, 0, stream>>>(acc, sbuf, bias0 + t * HID,
                                                  hA + DSTt[t] * NELEM);
  }
  relu_kernel<<<gAll3, b256, 0, stream>>>(hA, 3 * NELEM);

  // ---------------- layer 1 (H=64 -> H=64) ----------------
  zero_kernel<<<gAll3, b256, 0, stream>>>(hB, 3 * NELEM);
  for (int t = 0; t < 7; ++t) {
    const int* esrc = edges + t * 2 * EDGES;
    const int* edst = esrc + EDGES;
    gemm_node_kernel<<<gGemm, 128, 0, stream>>>(
        hA + SRCt[t] * NELEM, Wl1 + t * HID * HID, bl1 + t * HID, xl, HID);
    gemm_node_kernel<<<gGemm, 128, 0, stream>>>(
        hA + DSTt[t] * NELEM, Wr1 + t * HID * HID, br1 + t * HID, xr, HID);
    seg_init_kernel<<<gNode, b256, 0, stream>>>(menc, sbuf, acc);
    edge_attn_kernel<<<gEdge, b256, 0, stream>>>(esrc, edst, xl, xr,
                                                 att1 + t * HID, ebuf, menc);
    edge_scatter_kernel<<<gEdge, b256, 0, stream>>>(esrc, edst, xl, ebuf, menc,
                                                    sbuf, acc);
    node_accum_kernel<<<gNode, b256, 0, stream>>>(acc, sbuf, bias1 + t * HID,
                                                  hB + DSTt[t] * NELEM);
  }
  relu_kernel<<<gAll3, b256, 0, stream>>>(hB, 3 * NELEM);

  // ---------------- output head on glomeruli (type 0) ----------------
  head_kernel<<<(NNODES + 255) / 256, b256, 0, stream>>>(hB, Wout, bout, out);
}
